// MultiHeadAttentionQuantum_65481071400842
// MI455X (gfx1250) — matrix operand order repaired and checked by reading, not compiled, not run on hardware
//
#include <hip/hip_runtime.h>

typedef _Float16     v16h  __attribute__((ext_vector_type(16)));
typedef float        v8f   __attribute__((ext_vector_type(8)));
typedef unsigned int u32;
typedef u32          u32x4 __attribute__((ext_vector_type(4)));
typedef u32          u32x8 __attribute__((ext_vector_type(8)));

#define BATCH 4
#define SEQ   2048
#define EMB   512
#define HEADS 8
#define DK    64
#define ROWS  (BATCH * SEQ)          // 8192
#define KV_STRIDE 72                 // 64 halves + 8 halves pad (TDM-applied)
#define W_STRIDE  40                 // 32 halves + 8 halves pad (TDM-applied)
#define NCHUNK (SEQ / 32)            // 64 key chunks
// softmax constants: p = exp2(s_raw*C1 - C2) == exp((s_raw/8) - 8), exact shift
#define C1 0.1803368801f             // 0.125 * log2(e)
#define C2 11.5415603281f            // 8.0   * log2(e)

// ---------------------------------------------------------------------------
// WMMA helper: D = A(16x32 f16) * B(32x16 f16) + C(16x16 f32)
// ---------------------------------------------------------------------------
__device__ __forceinline__ v8f wmma_f16(v16h a, v16h b, v8f c) {
  return __builtin_amdgcn_wmma_f32_16x16x32_f16(
      /*neg_a=*/false, a, /*neg_b=*/false, b,
      /*c_mod=*/(short)0, c, /*reuse_a=*/false, /*reuse_b=*/false);
}

// A-fragment (16x32 f16): row = lane%16; elem e -> K = (lane/16)*16 + e
__device__ __forceinline__ v16h load_afrag(const _Float16* p, int ld, int lane) {
  union { v16h v; _Float16 h[16]; } u;
  int r = lane & 15, c = lane >> 4;
  const _Float16* rp = p + (long)r * ld + c * 16;
#pragma unroll
  for (int hh = 0; hh < 2; ++hh)
#pragma unroll
    for (int j = 0; j < 8; ++j)
      u.h[hh * 8 + j] = rp[hh * 8 + j];
  return u.v;
}

// B-fragment (32x16 f16) from B^T stored row-major: col n = lane%16; K = (lane/16)*16 + e
__device__ __forceinline__ v16h load_bfrag(const _Float16* p, int ld, int lane) {
  union { v16h v; _Float16 h[16]; } u;
  int n = lane & 15, c = lane >> 4;
  const _Float16* rp = p + (long)n * ld + c * 16;
#pragma unroll
  for (int j = 0; j < 16; ++j) u.h[j] = rp[j];
  return u.v;
}

// ---------------------------------------------------------------------------
// TDM descriptors (ISA ch.8): 2D tile, optional LDS row padding.
// ds_code: 0=1B 1=2B 2=4B. pad_interval k -> pad after 2^(k+1) DWORDs;
// pad_amount a -> a+1 DWORDs. Dims/strides in elements.
// ---------------------------------------------------------------------------
__device__ __forceinline__ void tdm_desc(u32x4& g0, u32x8& g1,
                                         const void* gaddr, u32 lds_off, u32 ds_code,
                                         u32 width, u32 rows, u32 src_stride,
                                         u32 pad_en, u32 pad_iv, u32 pad_am) {
  unsigned long long ga = (unsigned long long)(uintptr_t)gaddr;
  g0[0] = 1u;                                          // count=1, user D#
  g0[1] = lds_off;                                     // lds_addr (bytes)
  g0[2] = (u32)ga;                                     // global_addr[31:0]
  g0[3] = (u32)((ga >> 32) & 0x1FFFFFFu) | (2u << 30); // addr[56:32] | type=2
  g1[0] = (ds_code << 16) | (pad_en << 20) | (pad_iv << 22) | (pad_am << 25);
  g1[1] = width << 16;      // tensor_dim0
  g1[2] = rows << 16;       // tensor_dim1
  g1[3] = width << 16;      // tile_dim0
  g1[4] = rows;             // tile_dim1 (tile_dim2 = 0)
  g1[5] = src_stride;       // tensor_dim0_stride
  g1[6] = 0u;
  g1[7] = 0u;
}

__device__ __forceinline__ void tdm_load_f16(const _Float16* gsrc, u32 lds_off,
                                             u32 width, u32 rows, u32 stride,
                                             u32 pad_iv, u32 pad_am) {
  u32x4 g0; u32x8 g1;
  tdm_desc(g0, g1, gsrc, lds_off, 1u, width, rows, stride, 1u, pad_iv, pad_am);
  asm volatile("tensor_load_to_lds %0, %1" :: "s"(g0), "s"(g1) : "memory");
}

__device__ __forceinline__ void tdm_store_f16(_Float16* gdst, u32 lds_off,
                                              u32 width, u32 rows, u32 stride) {
  u32x4 g0; u32x8 g1;
  tdm_desc(g0, g1, gdst, lds_off, 1u, width, rows, stride, 0u, 0u, 0u);
  asm volatile("tensor_store_from_lds %0, %1" :: "s"(g0), "s"(g1) : "memory");
}

__device__ __forceinline__ void tdm_store_f32(float* gdst, u32 lds_off,
                                              u32 width, u32 rows, u32 stride) {
  u32x4 g0; u32x8 g1;
  tdm_desc(g0, g1, gdst, lds_off, 2u, width, rows, stride, 0u, 0u, 0u);
  asm volatile("tensor_store_from_lds %0, %1" :: "s"(g0), "s"(g1) : "memory");
}

// ---------------------------------------------------------------------------
// Kernel 1: e = cumprod(cos(x)) per row of 512, stored f16.
// ---------------------------------------------------------------------------
__global__ __launch_bounds__(128) void encode_kernel(const float* __restrict__ x,
                                                     _Float16* __restrict__ e16) {
  int lane = threadIdx.x & 31;
  int wave = threadIdx.x >> 5;
  long row = (long)blockIdx.x * 4 + wave;
  if (row >= ROWS) return;

  const float* xp = x + row * EMB + lane * 16;
  float pre[16];
  float prod = 1.f;
#pragma unroll
  for (int i = 0; i < 16; ++i) {
    prod *= __cosf(xp[i]);
    pre[i] = prod;
  }
  float incl = prod;
#pragma unroll
  for (int off = 1; off < 32; off <<= 1) {
    float t = __shfl_up(incl, off, 32);
    if (lane >= off) incl *= t;
  }
  float excl = __shfl_up(incl, 1, 32);
  if (lane == 0) excl = 1.f;

  _Float16* op = e16 + row * EMB + lane * 16;
#pragma unroll
  for (int i = 0; i < 16; ++i) op[i] = (_Float16)(excl * pre[i]);
}

// ---------------------------------------------------------------------------
// Kernel 2: convert W_out to f16.
// ---------------------------------------------------------------------------
__global__ void wconv_kernel(const float* __restrict__ w, _Float16* __restrict__ w16) {
  int i = blockIdx.x * blockDim.x + threadIdx.x;
  if (i < EMB * EMB) w16[i] = (_Float16)w[i];
}

// ---------------------------------------------------------------------------
// Kernel 3: flash attention per (b,h). Block = 4 waves = 64 queries.
// Scores bounded by |q.k|/8 <= 8, so softmax uses an exact constant shift
// (no running max); the denominator is accumulated by an extra WMMA against
// an all-ones B block. K/V tiles DMA'd in by TDM; O DMA'd out by TDM.
// exp args are in [-23, 0] -> raw v_exp_f32 (no denormal fixup needed).
// ---------------------------------------------------------------------------
__global__ __launch_bounds__(128) void attn_kernel(const _Float16* __restrict__ e16,
                                                   _Float16* __restrict__ pre16) {
  __shared__ _Float16 kv[2][32 * KV_STRIDE];  // double-buffered 32x64 tiles
  __shared__ _Float16 pt[4][32 * 16];         // per-wave P^T staging
  __shared__ _Float16 ost[4][16 * 64];        // per-wave O staging for TDM store

  int lane = threadIdx.x & 31;
  int wave = threadIdx.x >> 5;
  int blk  = blockIdx.x;                 // B*H*(S/64) = 1024
  int qb   = blk & 31;
  int bh   = blk >> 5;
  int h    = bh & 7, b = bh >> 3;

  const _Float16* base = e16 + ((long)b * SEQ) * EMB + h * DK;  // [S][64], ld=512
  int q0 = qb * 64 + wave * 16;

  v16h aq0 = load_afrag(base + (long)q0 * EMB + 0,  EMB, lane);
  v16h aq1 = load_afrag(base + (long)q0 * EMB + 32, EMB, lane);

  union { v16h v; _Float16 h[16]; } ones;
#pragma unroll
  for (int i = 0; i < 16; ++i) ones.h[i] = (_Float16)1.0f;

  v8f o[4] = {v8f{}, v8f{}, v8f{}, v8f{}};
  v8f lacc = {};                          // softmax denominators (all columns)

  // prologue: DMA tile 0 into buffer 0 (TENSORcnt is per-wave; wave 0 drives)
  if (wave == 0)
    tdm_load_f16(base, (u32)(uintptr_t)&kv[0][0], DK, 32, EMB, 4u, 3u);

  for (int kc = 0; kc < NCHUNK; ++kc) {
    int buf = kc & 1;
    __syncthreads();   // everyone done reading kv[1-buf] from previous iter
    if (wave == 0) {
      if (kc + 1 < NCHUNK) {
        tdm_load_f16(base + (long)((kc + 1) * 32) * EMB,
                     (u32)(uintptr_t)&kv[1 - buf][0], DK, 32, EMB, 4u, 3u);
        __builtin_amdgcn_s_wait_tensorcnt(1);   // tile kc complete
      } else {
        __builtin_amdgcn_s_wait_tensorcnt(0);
      }
    }
    __syncthreads();   // kv[buf] visible to all waves
    const _Float16* kvb = kv[buf];

    // scores -> p = exp2(s*C1 - C2), exact softmax numerator
    v8f s[2];
#pragma unroll
    for (int t = 0; t < 2; ++t) {
      v16h b0 = load_bfrag(&kvb[(t * 16) * KV_STRIDE + 0],  KV_STRIDE, lane);
      v16h b1 = load_bfrag(&kvb[(t * 16) * KV_STRIDE + 32], KV_STRIDE, lane);
      v8f z = {};
      z = wmma_f16(aq0, b0, z);
      z = wmma_f16(aq1, b1, z);
#pragma unroll
      for (int v = 0; v < 8; ++v)
        z[v] = __builtin_amdgcn_exp2f(z[v] * C1 - C2);   // bare v_exp_f32
      s[t] = z;
    }

    // stage P^T[k][q] (f16) to per-wave LDS (same-wave producer/consumer:
    // LDS ops are in-order within a wave; wave_barrier stops reordering)
#pragma unroll
    for (int t = 0; t < 2; ++t) {
      int krow = t * 16 + (lane & 15);
      int qcol = 8 * (lane >> 4);
      _Float16* dst = &pt[wave][krow * 16 + qcol];
#pragma unroll
      for (int v = 0; v < 8; ++v) dst[v] = (_Float16)s[t][v];
    }
    __builtin_amdgcn_wave_barrier();

    // P as A-fragment (16 queries x 32 keys)
    union { v16h v; _Float16 hh[16]; } ap;
    {
      int q = lane & 15, c = lane >> 4;
#pragma unroll
      for (int e = 0; e < 16; ++e) {
        int K = c * 16 + e;
        ap.hh[e] = pt[wave][K * 16 + q];
      }
    }
    // O += P * V ; denominators += P * ones
#pragma unroll
    for (int nb = 0; nb < 4; ++nb) {
      union { v16h v; _Float16 hh[16]; } bv;
      int n = nb * 16 + (lane & 15), c = lane >> 4;
#pragma unroll
      for (int j = 0; j < 16; ++j) bv.hh[j] = kvb[(c * 16 + j) * KV_STRIDE + n];
      o[nb] = wmma_f16(ap.v, bv.v, o[nb]);
    }
    lacc = wmma_f16(ap.v, ones.v, lacc);
  }

  // normalize, stage dense 16x64 O tile in LDS, DMA to global via TDM store
  int r = lane & 15, c = lane >> 4;
#pragma unroll
  for (int v = 0; v < 8; ++v) {
    float inv = 1.0f / lacc[v];
#pragma unroll
    for (int nb = 0; nb < 4; ++nb)
      ost[wave][(v + 8 * c) * 64 + nb * 16 + r] = (_Float16)(o[nb][v] * inv);
  }
  asm volatile("s_wait_dscnt 0x0" ::: "memory");   // LDS writes done before DMA
  tdm_store_f16(pre16 + ((long)b * SEQ + q0) * EMB + h * DK,
                (u32)(uintptr_t)&ost[wave][0], DK, 16, EMB);
  __builtin_amdgcn_s_wait_tensorcnt(0);
}

// ---------------------------------------------------------------------------
// Kernel 4: out[8192x512] = pre16 @ W^T, f32 out.
// Block = 8 waves sharing one 64-col W^T panel staged in LDS by the TDM
// (double-buffered); each wave owns a 16-row A stream -> 16x64 output,
// written back through tensor_store_from_lds (4B elements).
// ---------------------------------------------------------------------------
__global__ __launch_bounds__(256) void gemm_kernel(const _Float16* __restrict__ pre16,
                                                   const _Float16* __restrict__ w16,
                                                   float* __restrict__ out) {
  __shared__ _Float16 wt[2][64 * W_STRIDE];   // 64 W rows x 32 k, padded stride
  __shared__ float    ostg[8][16 * 64];       // per-wave f32 output staging

  int lane = threadIdx.x & 31;
  int wave = threadIdx.x >> 5;
  int tj   = blockIdx.x & 7;                  // 8 column groups of 64
  int tig  = blockIdx.x >> 3;                 // 64 groups of 8 row tiles
  long i0 = ((long)tig * 8 + wave) * 16;
  long j0 = (long)tj * 64;

  v8f acc[4] = {v8f{}, v8f{}, v8f{}, v8f{}};

  if (wave == 0)   // W^T panel, k-chunk 0: rows j0..j0+63, 32 halves/row
    tdm_load_f16(w16 + j0 * EMB, (u32)(uintptr_t)&wt[0][0], 32, 64, EMB, 3u, 3u);

  for (int kc = 0; kc < EMB / 32; ++kc) {
    int buf = kc & 1;
    __syncthreads();
    if (wave == 0) {
      if (kc + 1 < EMB / 32) {
        tdm_load_f16(w16 + j0 * EMB + (kc + 1) * 32,
                     (u32)(uintptr_t)&wt[1 - buf][0], 32, 64, EMB, 3u, 3u);
        __builtin_amdgcn_s_wait_tensorcnt(1);
      } else {
        __builtin_amdgcn_s_wait_tensorcnt(0);
      }
    }
    __syncthreads();

    __builtin_prefetch(pre16 + i0 * EMB + (kc + 2) * 32, 0, 1);
    v16h a = load_afrag(pre16 + i0 * EMB + kc * 32, EMB, lane);
#pragma unroll
    for (int jb = 0; jb < 4; ++jb) {
      v16h b = load_bfrag(&wt[buf][(jb * 16) * W_STRIDE], W_STRIDE, lane);
      acc[jb] = wmma_f16(a, b, acc[jb]);
    }
  }

  // stage 16x64 f32 tile densely in LDS, DMA out via TDM
  int r = lane & 15, c = lane >> 4;
#pragma unroll
  for (int jb = 0; jb < 4; ++jb)
#pragma unroll
    for (int v = 0; v < 8; ++v)
      ostg[wave][(v + 8 * c) * 64 + jb * 16 + r] = acc[jb][v];
  asm volatile("s_wait_dscnt 0x0" ::: "memory");
  tdm_store_f32(out + i0 * EMB + j0, (u32)(uintptr_t)&ostg[wave][0], 64, 16, EMB);
  __builtin_amdgcn_s_wait_tensorcnt(0);
}

// ---------------------------------------------------------------------------
extern "C" void kernel_launch(void* const* d_in, const int* in_sizes, int n_in,
                              void* d_out, int out_size, void* d_ws, size_t ws_size,
                              hipStream_t stream) {
  const float* x = (const float*)d_in[0];   // [4,2048,512] f32
  const float* W = (const float*)d_in[1];   // [512,512] f32
  float* out = (float*)d_out;               // [4,2048,512] f32

  char* ws = (char*)d_ws;
  const size_t E16_BYTES = (size_t)ROWS * EMB * sizeof(_Float16);   // 8 MB
  _Float16* e16 = (_Float16*)ws;
  _Float16* pre = (_Float16*)(ws + E16_BYTES);
  _Float16* w16 = (_Float16*)(ws + 2 * E16_BYTES);

  encode_kernel<<<ROWS / 4, 128, 0, stream>>>(x, e16);
  wconv_kernel<<<(EMB * EMB) / 256, 256, 0, stream>>>(W, w16);
  attn_kernel<<<BATCH * HEADS * (SEQ / 64), 128, 0, stream>>>(e16, pre);
  gemm_kernel<<<(EMB / 64) * (ROWS / 16 / 8), 256, 0, stream>>>(pre, w16, out);
}
